// EmbedNode_31963146617220
// MI455X (gfx1250) — compile-verified, matrix operand order checked by reading, not run on hardware
//
#include <hip/hip_runtime.h>
#include <stdint.h>

typedef __attribute__((ext_vector_type(16))) _Float16 v16h;
typedef __attribute__((ext_vector_type(8)))  float    v8f;

#define N_NODES   10000
#define E_TOT     160000
#define WN        2304          // 32*32 + 32*16 + 16*16 + 16*32
#define NCHUNK    144           // WN / 16
#define WAVES     8             // waves per workgroup (128 edges/WG)
#define INV_SQRT3 0.57735026918962576f
#define ALPHA     0.14433756729740643f   // 1/sqrt(48)
#define INV_SQRT32 0.17677669529663687f
#define EPSV      1e-5f

// ---------------- workspace byte offsets ----------------
#define OFF_W1     0u                       // 8192 halfs  (16 KB)
#define OFF_W2     16384u                   // 294912 halfs (576 KB)
#define OFF_SUMS   606208u                  // N*80 f32
#define OFF_CNT    3806208u                 // N f32
#define OFF_STATS  3846208u                 // 80 f32 (sum_s, sumsq_s, q)
#define OFF_SCALES 3846528u                 // 80 f32 (m, ks, kv)

// ---------------- CDNA5 async copy helpers ----------------
__device__ __forceinline__ void async_copy_16B(uint32_t lds_off, const void* gbase, uint32_t goff) {
    // GVS mode: mem = SADDR + VADDR + IOFFSET ; LDS dst = VDST + IOFFSET
    asm volatile("global_load_async_to_lds_b128 %0, %1, %2"
                 :: "v"(lds_off), "v"(goff), "s"(gbase)
                 : "memory");
}
#define ASYNC_WAIT_1() asm volatile("s_wait_asynccnt 0x1" ::: "memory")
#define ASYNC_WAIT_0() asm volatile("s_wait_asynccnt 0x0" ::: "memory")

// =========================================================
// Kernel 1: convert W1/W2 to f16 in B-operand order; zero accumulators
// B-layout (f16 KxN, ISA 7.12.2): lane l -> n = l%16,
// VGPR v holds K pair at kbase = 16*(v>=4) + 8*(l>=16) + 2*(v%4).
// =========================================================
__global__ void prep_kernel(const float* __restrict__ fw1, const float* __restrict__ fw2,
                            _Float16* __restrict__ w1h, _Float16* __restrict__ w2h,
                            float* __restrict__ sums, float* __restrict__ cnt,
                            float* __restrict__ stats) {
    int t = blockIdx.x * 256 + threadIdx.x;
    if (t < 294912) {                                    // W2: 144 chunks * 4 ksteps * 32 lanes * 16 halfs
        int j = t & 15; int q = t >> 4; int l = q & 31; int qc = q >> 5;
        int s = qc & 3; int c = qc >> 2;
        int v = j >> 1;
        int k = s * 32 + ((v >= 4) ? 16 : 0) + ((l >= 16) ? 8 : 0) + (v & 3) * 2 + (j & 1);
        int n = l & 15;
        w2h[t] = (_Float16)fw2[k * WN + c * 16 + n];
    } else if (t < 303104) {                             // W1: 8 chunks * 2 ksteps * 32 * 16
        int t2 = t - 294912;
        int j = t2 & 15; int q = t2 >> 4; int l = q & 31; int qc = q >> 5;
        int s = qc & 1; int c = qc >> 1;
        int v = j >> 1;
        int k = s * 32 + ((v >= 4) ? 16 : 0) + ((l >= 16) ? 8 : 0) + (v & 3) * 2 + (j & 1);
        int n = l & 15;
        w1h[t2] = (_Float16)fw1[k * 128 + c * 16 + n];
    } else if (t < 1103104) { sums[t - 303104] = 0.0f; }
    else if (t < 1113104)   { cnt[t - 1103104] = 0.0f; }
    else if (t < 1113184)   { stats[t - 1113104] = 0.0f; }
}

// =========================================================
// Kernel 2: fused edge kernel. 8 waves/WG, 16 edges/wave.
// GEMM1 (16x64x128) + relu -> h ; GEMM2 (16x128x2304) chunked, each
// 16x16 w-chunk consumed immediately by the tensor-product contraction.
// W2 chunks streamed to LDS, triple-buffered async-to-LDS, 1 barrier/chunk.
// =========================================================
__global__ void __launch_bounds__(256)
edge_kernel(const float* __restrict__ nf, const float* __restrict__ ef,
            const float* __restrict__ esh, const int* __restrict__ eidx,
            const float* __restrict__ fb1, const float* __restrict__ fb2,
            const _Float16* __restrict__ w1h, const _Float16* __restrict__ w2h,
            float* __restrict__ sums, float* __restrict__ cnt) {
    __shared__ __attribute__((aligned(32))) _Float16 s_w2buf[3][2048]; // 12 KB triple buffer
    __shared__ float s_a00[WAVES][16][32];
    __shared__ float s_a01[WAVES][16][32];
    __shared__ float s_a11[WAVES][16][16];
    __shared__ float s_a10[WAVES][16][48];
    __shared__ float s_shv[WAVES][16][3];
    __shared__ int   s_src[WAVES][16];
    __shared__ __attribute__((aligned(4))) _Float16 s_h[WAVES][2048];  // 16x128 f16 per wave

    const int tid  = threadIdx.x;
    const int wv   = tid >> 5;
    const int lane = tid & 31;
    const int m    = lane & 15;
    const int half = lane >> 4;
    const int col  = m;                     // N index within a 16-col chunk
    const int e_base = blockIdx.x * (WAVES * 16) + wv * 16;
    const int e = e_base + m;

    // ---- per-edge TP coefficient tables ----
    const int dstn = eidx[e];
    const int srcn = eidx[E_TOT + e];
    const float sh0 = esh[e * 4 + 0], sh1 = esh[e * 4 + 1],
                sh2 = esh[e * 4 + 2], sh3 = esh[e * 4 + 3];
    if (half == 0) {
        s_src[wv][m] = srcn;
        s_shv[wv][m][0] = sh1; s_shv[wv][m][1] = sh2; s_shv[wv][m][2] = sh3;
    }
    const float* __restrict__ x = nf + (size_t)dstn * 80;
#pragma unroll
    for (int j = 0; j < 16; ++j) {
        int u = half * 16 + j;
        float v = x[u];
        s_a01[wv][m][u] = v;
        s_a00[wv][m][u] = v * sh0;
    }
#pragma unroll
    for (int j = 0; j < 8; ++j) {
        int u = half * 8 + j;
        float v0 = x[32 + u * 3 + 0], v1 = x[32 + u * 3 + 1], v2 = x[32 + u * 3 + 2];
        s_a10[wv][m][u * 3 + 0] = v0 * sh0;
        s_a10[wv][m][u * 3 + 1] = v1 * sh0;
        s_a10[wv][m][u * 3 + 2] = v2 * sh0;
        s_a11[wv][m][u] = (v0 * sh1 + v1 * sh2 + v2 * sh3) * INV_SQRT3;
    }

    // ---- GEMM1 A operand (edge_features f32 -> f16, ISA A-layout) ----
    const float2* __restrict__ efp = (const float2*)(ef + (size_t)e * 64);
    v16h a1[2];
#pragma unroll
    for (int s = 0; s < 2; ++s)
#pragma unroll
        for (int v = 0; v < 8; ++v) {
            int kb = s * 32 + ((v >= 4) ? 16 : 0) + half * 8 + (v & 3) * 2;
            float2 p = efp[kb >> 1];
            a1[s][2 * v]     = (_Float16)p.x;
            a1[s][2 * v + 1] = (_Float16)p.y;
        }

    // ---- GEMM1 + bias + relu -> h (LDS, row-major f16) ----
#pragma unroll
    for (int c2 = 0; c2 < 8; ++c2) {
        v8f acc = {};
#pragma unroll
        for (int s = 0; s < 2; ++s) {
            v16h b = *(const v16h*)(w1h + (((c2 * 2 + s) * 32 + lane) << 4));
            acc = __builtin_amdgcn_wmma_f32_16x16x32_f16(false, a1[s], false, b,
                                                         (short)0, acc, false, false);
        }
        float bb = fb1[c2 * 16 + col];
#pragma unroll
        for (int r = 0; r < 8; ++r) {
            float v = acc[r] + bb;
            v = v > 0.0f ? v : 0.0f;
            s_h[wv][(r + 8 * half) * 128 + c2 * 16 + col] = (_Float16)v;
        }
    }

    // ---- GEMM2 A operand from h (LDS -> ISA A-layout f16) ----
    v16h a2[4];
#pragma unroll
    for (int s2 = 0; s2 < 4; ++s2)
#pragma unroll
        for (int v = 0; v < 8; ++v) {
            int kb = s2 * 32 + ((v >= 4) ? 16 : 0) + half * 8 + (v & 3) * 2;
            union { uint32_t u32; _Float16 h[2]; } cv;
            cv.u32 = *(const uint32_t*)&s_h[wv][m * 128 + kb];
            a2[s2][2 * v]     = cv.h[0];
            a2[s2][2 * v + 1] = cv.h[1];
        }

    // ---- TP accumulators (match D-tile layout: reg r <-> edge r+8*half, lane%16 = col) ----
    float outs0[8], outs1[8], t01a[8], ov0[8], ov1[8], ov2[8];
#pragma unroll
    for (int r = 0; r < 8; ++r) { outs0[r] = 0; outs1[r] = 0; t01a[r] = 0; ov0[r] = 0; ov1[r] = 0; ov2[r] = 0; }

    // ---- chunk loop: triple-buffered async W2 streaming, 1 barrier/chunk ----
    const uint32_t lds0 = (uint32_t)(uintptr_t)(&s_w2buf[0][0]) + (uint32_t)tid * 16u;
    async_copy_16B(lds0,         w2h, (uint32_t)tid * 16u);            // chunk 0 -> buf 0
    async_copy_16B(lds0 + 4096u, w2h, 4096u + (uint32_t)tid * 16u);    // chunk 1 -> buf 1

    for (int c = 0; c < NCHUNK; ++c) {
        if (c + 1 < NCHUNK) { ASYNC_WAIT_1(); } else { ASYNC_WAIT_0(); }
        __syncthreads();

        const _Float16* __restrict__ bbase = &s_w2buf[c % 3][0];
        v8f d = {};
#pragma unroll
        for (int s2 = 0; s2 < 4; ++s2) {
            v16h b = *(const v16h*)(bbase + ((s2 * 32 + lane) << 4));
            d = __builtin_amdgcn_wmma_f32_16x16x32_f16(false, a2[s2], false, b,
                                                       (short)0, d, false, false);
        }
        const float bias2 = fb2[c * 16 + col];

        if (c < 64) {                       // w00: out_s[(c&1)*16+n] += a00[u]*w
            const int u = c >> 1;
            if ((c & 1) == 0) {
#pragma unroll
                for (int r = 0; r < 8; ++r) outs0[r] += s_a00[wv][8 * half + r][u] * (d[r] + bias2);
            } else {
#pragma unroll
                for (int r = 0; r < 8; ++r) outs1[r] += s_a00[wv][8 * half + r][u] * (d[r] + bias2);
            }
        } else if (c < 96) {                // w01: t01[n] += xs[u]*w
            const int u = c - 64;
#pragma unroll
            for (int r = 0; r < 8; ++r) t01a[r] += s_a01[wv][8 * half + r][u] * (d[r] + bias2);
        } else if (c < 112) {               // w10: ov[n][i] += xv[u][i]*sh_s*w
            const int u = c - 96;
#pragma unroll
            for (int r = 0; r < 8; ++r) {
                float v = d[r] + bias2;
                ov0[r] += s_a10[wv][8 * half + r][u * 3 + 0] * v;
                ov1[r] += s_a10[wv][8 * half + r][u * 3 + 1] * v;
                ov2[r] += s_a10[wv][8 * half + r][u * 3 + 2] * v;
            }
        } else {                            // w11: out_s += a11[u]*w
            const int cc = c - 112;
            const int u = cc >> 1;
            if ((cc & 1) == 0) {
#pragma unroll
                for (int r = 0; r < 8; ++r) outs0[r] += s_a11[wv][8 * half + r][u] * (d[r] + bias2);
            } else {
#pragma unroll
                for (int r = 0; r < 8; ++r) outs1[r] += s_a11[wv][8 * half + r][u] * (d[r] + bias2);
            }
        }

        if (c + 2 < NCHUNK) {               // prefetch chunk c+2 (safe: all waves past barrier c)
            const uint32_t nb = (uint32_t)((c + 2) % 3);
            async_copy_16B(lds0 + nb * 4096u, w2h,
                           (uint32_t)(c + 2) * 4096u + (uint32_t)tid * 16u);
        }
    }

    // ---- scatter: tp * alpha atomically aggregated by edge_src ----
#pragma unroll
    for (int r = 0; r < 8; ++r) {
        const int me = 8 * half + r;
        const int sn = s_src[wv][me];
        float* __restrict__ srow = sums + (size_t)sn * 80;
        atomicAdd(srow + col,      outs0[r] * ALPHA);
        atomicAdd(srow + 16 + col, outs1[r] * ALPHA);
        const float t = t01a[r];
        const float w0 = s_shv[wv][me][0], w1 = s_shv[wv][me][1], w2 = s_shv[wv][me][2];
        atomicAdd(srow + 32 + col * 3 + 0, (t * w0 + ov0[r]) * ALPHA);
        atomicAdd(srow + 32 + col * 3 + 1, (t * w1 + ov1[r]) * ALPHA);
        atomicAdd(srow + 32 + col * 3 + 2, (t * w2 + ov2[r]) * ALPHA);
    }
    if (lane < 16) atomicAdd(&cnt[s_src[wv][lane]], 1.0f);
}

// =========================================================
// Kernel 3: per-node agg/cnt + bypass matmuls, write pre-BN y to d_out,
// accumulate BN statistics (LDS reduce -> global atomics).
// =========================================================
__global__ void __launch_bounds__(256)
node_kernel(const float* __restrict__ sums, const float* __restrict__ cnt,
            const float* __restrict__ nf, const float* __restrict__ bws,
            const float* __restrict__ bwv, float* __restrict__ y,
            float* __restrict__ stats) {
    __shared__ float sws[1024];
    __shared__ float swv[256];
    __shared__ float ls[32], lss[32], lq[16];
    const int tid = threadIdx.x;
    for (int i = tid; i < 1024; i += 256) sws[i] = bws[i];
    if (tid < 256) swv[tid] = bwv[tid];
    if (tid < 32) { ls[tid] = 0.0f; lss[tid] = 0.0f; }
    if (tid < 16) lq[tid] = 0.0f;
    __syncthreads();

    const int n = blockIdx.x * 256 + tid;
    if (n < N_NODES) {
        const float inv = 1.0f / fmaxf(cnt[n], 1.0f);
        float xr[80];
#pragma unroll
        for (int i = 0; i < 80; ++i) xr[i] = nf[(size_t)n * 80 + i];
        for (int w = 0; w < 32; ++w) {
            float b = 0.0f;
#pragma unroll
            for (int u = 0; u < 32; ++u) b += xr[u] * sws[u * 32 + w];
            float v = sums[(size_t)n * 80 + w] * inv + b * INV_SQRT32;
            y[(size_t)n * 80 + w] = v;
            atomicAdd(&ls[w], v);
            atomicAdd(&lss[w], v * v);
        }
        for (int w = 0; w < 16; ++w) {
            float q = 0.0f;
#pragma unroll
            for (int i = 0; i < 3; ++i) {
                float b = 0.0f;
#pragma unroll
                for (int u = 0; u < 16; ++u) b += xr[32 + u * 3 + i] * swv[u * 16 + w];
                float v = sums[(size_t)n * 80 + 32 + w * 3 + i] * inv + b * 0.25f;
                y[(size_t)n * 80 + 32 + w * 3 + i] = v;
                q += v * v;
            }
            atomicAdd(&lq[w], q * (1.0f / 3.0f));
        }
    }
    __syncthreads();
    if (tid < 32) { atomicAdd(&stats[tid], ls[tid]); atomicAdd(&stats[32 + tid], lss[tid]); }
    if (tid < 16) atomicAdd(&stats[64 + tid], lq[tid]);
}

// =========================================================
// Kernel 4: finalize BN scales
// =========================================================
__global__ void stats_kernel(const float* __restrict__ stats, const float* __restrict__ bnws,
                             const float* __restrict__ bnwv, float* __restrict__ scales) {
    const int t = threadIdx.x;
    const float invN = 1.0f / (float)N_NODES;
    if (t < 32) {
        float mn = stats[t] * invN;
        float var = stats[32 + t] * invN - mn * mn;
        scales[t] = mn;
        scales[32 + t] = bnws[t] * rsqrtf(var + EPSV);
    } else if (t < 48) {
        int w = t - 32;
        scales[64 + w] = bnwv[w] * rsqrtf(stats[64 + w] * invN + EPSV);
    }
}

// =========================================================
// Kernel 5: apply BN in place on d_out
// =========================================================
__global__ void apply_kernel(float* __restrict__ y, const float* __restrict__ scales,
                             const float* __restrict__ bnbs) {
    const int t = blockIdx.x * 256 + threadIdx.x;
    if (t >= N_NODES * 80) return;
    const int c = t % 80;
    float v = y[t];
    if (c < 32) v = (v - scales[c]) * scales[32 + c] + bnbs[c];
    else        v = v * scales[64 + (c - 32) / 3];
    y[t] = v;
}

// =========================================================
extern "C" void kernel_launch(void* const* d_in, const int* in_sizes, int n_in,
                              void* d_out, int out_size, void* d_ws, size_t ws_size,
                              hipStream_t stream) {
    const float* nf   = (const float*)d_in[0];
    const float* ef   = (const float*)d_in[1];
    const float* esh  = (const float*)d_in[2];
    const int*   eidx = (const int*)d_in[3];
    const float* fw1  = (const float*)d_in[4];
    const float* fb1  = (const float*)d_in[5];
    const float* fw2  = (const float*)d_in[6];
    const float* fb2  = (const float*)d_in[7];
    const float* bws  = (const float*)d_in[8];
    const float* bwv  = (const float*)d_in[9];
    const float* bnws = (const float*)d_in[10];
    const float* bnwv = (const float*)d_in[11];
    const float* bnbs = (const float*)d_in[12];
    float* out = (float*)d_out;

    char* ws = (char*)d_ws;
    _Float16* w1h   = (_Float16*)(ws + OFF_W1);
    _Float16* w2h   = (_Float16*)(ws + OFF_W2);
    float* sums     = (float*)(ws + OFF_SUMS);
    float* cnt      = (float*)(ws + OFF_CNT);
    float* stats    = (float*)(ws + OFF_STATS);
    float* scales   = (float*)(ws + OFF_SCALES);

    prep_kernel<<<4349, 256, 0, stream>>>(fw1, fw2, w1h, w2h, sums, cnt, stats);
    edge_kernel<<<E_TOT / (WAVES * 16), 256, 0, stream>>>(nf, ef, esh, eidx, fb1, fb2, w1h, w2h, sums, cnt);
    node_kernel<<<(N_NODES + 255) / 256, 256, 0, stream>>>(sums, cnt, nf, bws, bwv, out, stats);
    stats_kernel<<<1, 64, 0, stream>>>(stats, bnws, bnwv, scales);
    apply_kernel<<<(N_NODES * 80 + 255) / 256, 256, 0, stream>>>(out, scales, bnbs);
}
